// RNNImputer_61933428408498
// MI455X (gfx1250) — compile-verified, matrix operand order checked by reading, not run on hardware
//
#include <hip/hip_runtime.h>
#include <hip/hip_bf16.h>

// ---------------------------------------------------------------------------
// Masked-GRU imputer, MI455X (gfx1250).
// Latency-bound sequential scan: 32 independent workgroups (one 16-row batch
// tile each), 8 wave32s per WGP, wave j owns hidden-column block 16j..16j+15
// of all three gates (18 v_wmma_f32_16x16x32_bf16 per wave per step).
// Loop-invariant weight B-fragments in VGPRs, hidden state double-buffered
// through LDS as bf16, only 2 barriers per step: waves 0-1 own the entire
// serial tail (readout GEMM + imputation fill) column-wise, so the
// sXhat write->read dependency is wave-local (DScnt-ordered, no barrier).
// ---------------------------------------------------------------------------

typedef __bf16 bf16_t;
typedef __attribute__((ext_vector_type(16))) __bf16 v16bf;
typedef __attribute__((ext_vector_type(8)))  __bf16 v8bf;
typedef __attribute__((ext_vector_type(2)))  __bf16 v2bf;
typedef __attribute__((ext_vector_type(8)))  float  v8f;

#define WMMA_BF16(a, b, c) \
  __builtin_amdgcn_wmma_f32_16x16x32_bf16(false, (a), false, (b), (short)0, (c), false, false)

namespace {
constexpr int T_STEPS = 512;
constexpr int B_TOT   = 512;
constexpr int D_IN    = 32;
constexpr int DM      = 128;   // d_model
constexpr int DIN2    = 64;    // 2*D_IN (concat_mask)
constexpr int BM      = 16;    // batch rows per workgroup (WMMA M)
}

static __device__ __forceinline__ float fast_rcp(float v) {
#if __has_builtin(__builtin_amdgcn_rcpf)
  return __builtin_amdgcn_rcpf(v);     // v_rcp_f32
#else
  return 1.0f / v;
#endif
}
static __device__ __forceinline__ float fast_sigmoid(float v) {
  return fast_rcp(1.0f + __expf(-v));  // v_exp_f32 + v_add + v_rcp
}
static __device__ __forceinline__ float fast_tanh(float v) {
#if __has_builtin(__builtin_amdgcn_tanhf)
  return __builtin_amdgcn_tanhf(v);    // v_tanh_f32 (gfx1250 TRANS op)
#else
  return __builtin_fmaf(2.0f, fast_rcp(1.0f + __expf(-2.0f * v)), -1.0f);
#endif
}

// Load a 16x32 bf16 fragment (A-layout, or B-layout with lane->N) from LDS.
// 16-bit layout: lanes 0-15 -> row, K chunks {kb..kb+7} and {kb+16..kb+23};
// lanes 16-31 shift K base by +8.  Two ds_load_b128 per lane.
static __device__ __forceinline__ v16bf load_frag_lds(const bf16_t* base, int ldk,
                                                      int lane, int kbase) {
  const int m   = lane & 15;
  const int klo = kbase + ((lane & 16) >> 1);      // +8 for upper half-wave
  const bf16_t* p = base + m * ldk + klo;
  v8bf lo = *(const v8bf*)(p);
  v8bf hi = *(const v8bf*)(p + 16);
  return __builtin_shufflevector(lo, hi, 0, 1, 2, 3, 4, 5, 6, 7,
                                 8, 9, 10, 11, 12, 13, 14, 15);
}

// One-time: build a B-fragment (K x 16, lane -> output feature N) directly from
// the f32 weight matrix W[N][K] in global memory, converting to bf16.
static __device__ __forceinline__ v16bf load_bfrag_global(const float* __restrict__ W,
                                                          int ldk, int lane,
                                                          int nbase, int kbase) {
  const int n   = nbase + (lane & 15);
  const int klo = kbase + ((lane & 16) >> 1);
  const float* p = W + n * ldk + klo;
  v16bf r;
#pragma unroll
  for (int i = 0; i < 8; ++i) {
    r[i]     = (bf16_t)p[i];
    r[8 + i] = (bf16_t)p[16 + i];
  }
  return r;
}

__global__ __launch_bounds__(256, 1)
void rnn_imputer_kernel(const float* __restrict__ x,
                        const float* __restrict__ W_ih,
                        const float* __restrict__ W_hh,
                        const float* __restrict__ b_ih,
                        const float* __restrict__ b_hh,
                        const float* __restrict__ W_out,
                        const float* __restrict__ b_out,
                        const int*   __restrict__ mask,
                        float*       __restrict__ out) {
  __shared__ __align__(16) bf16_t sH[2][BM * DM];    // double-buffered h (bf16)
  __shared__ __align__(16) bf16_t sX[BM * DIN2];     // x_t (bf16)
  __shared__ __align__(16) bf16_t sWout[D_IN * DM];  // W_out (bf16)
  __shared__ __align__(8)  float  sXhat[BM * D_IN];  // x_hat (f32)

  const int tid  = threadIdx.x;
  const int lane = tid & 31;
  const int wave = tid >> 5;                          // 0..7 = hidden col block
  const int b0   = blockIdx.x * BM;

  // ---- loop-invariant weight fragments in VGPRs ----
  v16bf whh_r[4], whh_z[4], whh_n[4];                 // K=128 -> 4 frags each
#pragma unroll
  for (int kf = 0; kf < 4; ++kf) {
    whh_r[kf] = load_bfrag_global(W_hh, DM, lane, 16 * wave,          32 * kf);
    whh_z[kf] = load_bfrag_global(W_hh, DM, lane, DM + 16 * wave,     32 * kf);
    whh_n[kf] = load_bfrag_global(W_hh, DM, lane, 2 * DM + 16 * wave, 32 * kf);
  }
  v16bf wih_r[2], wih_z[2], wih_n[2];                 // K=64 -> 2 frags each
#pragma unroll
  for (int kf = 0; kf < 2; ++kf) {
    wih_r[kf] = load_bfrag_global(W_ih, DIN2, lane, 16 * wave,          32 * kf);
    wih_z[kf] = load_bfrag_global(W_ih, DIN2, lane, DM + 16 * wave,     32 * kf);
    wih_n[kf] = load_bfrag_global(W_ih, DIN2, lane, 2 * DM + 16 * wave, 32 * kf);
  }

  // Per-lane biases (column = output feature of this lane's N slot).
  const int   ncol    = 16 * wave + (lane & 15);
  const float bias_r  = b_ih[ncol]          + b_hh[ncol];
  const float bias_z  = b_ih[DM + ncol]     + b_hh[DM + ncol];
  const float bias_xn = b_ih[2 * DM + ncol];
  const float bias_hn = b_hh[2 * DM + ncol];
  const float bias_o  = (wave < 2) ? b_out[16 * wave + (lane & 15)] : 0.0f;

  // W_out f32 -> bf16 LDS copy (32x128, 16 elems/thread).
#pragma unroll
  for (int i = 0; i < 16; ++i) {
    const int idx = tid * 16 + i;
    sWout[idx] = (bf16_t)W_out[idx];
  }
  // h0 = 0 in buffer 0.
#pragma unroll
  for (int i = 0; i < 8; ++i) sH[0][tid * 8 + i] = (bf16_t)0.0f;

  // ---- one-time init: x_hat0 = b_out, out[:,0,:], x_0 (all 256 threads) ----
  {
    const int row = tid >> 4;                 // 0..15
    const int d0  = (2 * tid) & 31;           // even, pair {d0, d0+1}
    const float2 bo = *(const float2*)&b_out[d0];
    *(float2*)&sXhat[row * D_IN + d0] = bo;
    *(float2*)&out[((size_t)(b0 + row) * T_STEPS) * D_IN + d0] = bo;

    const int   goff = ((b0 + row) * T_STEPS) * D_IN + d0;
    const float2 xv = *(const float2*)(x + goff);
    const int2   mv = *(const int2*)(mask + goff);
    v2bf xp, mf;
    xp[0] = (bf16_t)(mv.x ? xv.x : bo.x);
    xp[1] = (bf16_t)(mv.y ? xv.y : bo.y);
    mf[0] = (bf16_t)(mv.x ? 1.0f : 0.0f);
    mf[1] = (bf16_t)(mv.y ? 1.0f : 0.0f);
    *(v2bf*)&sX[row * DIN2 + d0]        = xp;
    *(v2bf*)&sX[row * DIN2 + D_IN + d0] = mf;
  }

  // ---- serial-tail ownership (waves 0-1): 4 (row, d-pair) slots per lane ----
  // Wave w owns x_hat / x_t columns 16w..16w+15 (and mask cols 32+16w..).
  int goff4[4], sxoff4[4], shoff4[4];
#pragma unroll
  for (int j = 0; j < 4; ++j) {
    const int idx  = lane * 4 + j;            // 0..127
    const int brow = idx >> 3;                // 0..15
    const int bd0  = 16 * wave + 2 * (idx & 7);
    goff4[j]  = ((b0 + brow) * T_STEPS) * D_IN + bd0;
    sxoff4[j] = brow * DIN2 + bd0;
    shoff4[j] = brow * D_IN + bd0;
  }

  // Readout waves' output mapping (C layout: M = v + 8*(lane/16)).
  const int rcol = 16 * wave + (lane & 15);
  const int rb   = (lane >> 4) * 8;
  float* op = out + (size_t)(b0 + rb) * T_STEPS * D_IN + (size_t)D_IN + rcol;

  v8f hc;                                     // persistent h tile (C layout)
#pragma unroll
  for (int e = 0; e < 8; ++e) hc[e] = 0.0f;

  int tc = D_IN;                              // global column offset of t+1

  __syncthreads();

  // -------------------- sequential scan over time --------------------
  for (int t = 0; t < T_STEPS - 1; ++t) {
    const int p = t & 1;

    v8f accR, accZ, accXN, accHN;
#pragma unroll
    for (int e = 0; e < 8; ++e) {
      accR[e] = bias_r; accZ[e] = bias_z; accXN[e] = bias_xn; accHN[e] = bias_hn;
    }

    // gx contributions (K = 64)
#pragma unroll
    for (int kf = 0; kf < 2; ++kf) {
      v16bf a = load_frag_lds(sX, DIN2, lane, 32 * kf);
      accR  = WMMA_BF16(a, wih_r[kf], accR);
      accZ  = WMMA_BF16(a, wih_z[kf], accZ);
      accXN = WMMA_BF16(a, wih_n[kf], accXN);
    }
    // gh contributions (K = 128)
#pragma unroll
    for (int kf = 0; kf < 4; ++kf) {
      v16bf a = load_frag_lds(sH[p], DM, lane, 32 * kf);
      accR  = WMMA_BF16(a, whh_r[kf], accR);
      accZ  = WMMA_BF16(a, whh_z[kf], accZ);
      accHN = WMMA_BF16(a, whh_n[kf], accHN);
    }

    // Gates — wave-local (h_old tile lives in registers in C layout).
#pragma unroll
    for (int e = 0; e < 8; ++e) {
      const float r = fast_sigmoid(accR[e]);
      const float z = fast_sigmoid(accZ[e]);
      const float n = fast_tanh(__builtin_fmaf(r, accHN[e], accXN[e]));
      hc[e] = __builtin_fmaf(z, hc[e] - n, n);     // (1-z)*n + z*h
    }

    // h_{t+1} -> other buffer (nobody reads it until the barrier).
#pragma unroll
    for (int v = 0; v < 8; ++v)
      sH[1 - p][(rb + v) * DM + rcol] = (bf16_t)hc[v];

    // Waves 0-1: issue next column's x/mask loads now; latency hides under
    // the barrier + readout WMMAs.
    float2 xv4[4]; int2 mv4[4];
    if (wave < 2) {
#pragma unroll
      for (int j = 0; j < 4; ++j) {
        xv4[j] = *(const float2*)(x    + goff4[j] + tc);
        mv4[j] = *(const int2*)  (mask + goff4[j] + tc);
      }
    }

    __syncthreads();                              // h_{t+1} visible

    // Serial tail, waves 0-1 only (column-owned, no further barrier needed):
    //   x_hat_{t+1} = h_{t+1}·W_outᵀ + b_out, emit out[:, t+1, :],
    //   then x_{t+1} = [mask ? x : x_hat, mask_f] into sX.
    if (wave < 2) {
      v8f xo;
#pragma unroll
      for (int e = 0; e < 8; ++e) xo[e] = bias_o;
#pragma unroll
      for (int kf = 0; kf < 4; ++kf) {
        v16bf a = load_frag_lds(sH[1 - p], DM, lane, 32 * kf);
        v16bf b = load_frag_lds(sWout + 16 * wave * DM, DM, lane, 32 * kf);
        xo = WMMA_BF16(a, b, xo);
      }
#pragma unroll
      for (int v = 0; v < 8; ++v) {
        sXhat[(rb + v) * D_IN + rcol] = xo[v];
        op[(size_t)v * T_STEPS * D_IN] = xo[v];
      }
      op += D_IN;

      // Imputation fill: reads sXhat written above by this same wave
      // (same-wave LDS ordering via DScnt; no barrier required).
#pragma unroll
      for (int j = 0; j < 4; ++j) {
        const float2 xh = *(const float2*)&sXhat[shoff4[j]];
        v2bf xp2, mf2;
        xp2[0] = (bf16_t)(mv4[j].x ? xv4[j].x : xh.x);
        xp2[1] = (bf16_t)(mv4[j].y ? xv4[j].y : xh.y);
        mf2[0] = (bf16_t)(mv4[j].x ? 1.0f : 0.0f);
        mf2[1] = (bf16_t)(mv4[j].y ? 1.0f : 0.0f);
        *(v2bf*)&sX[sxoff4[j]]        = xp2;
        *(v2bf*)&sX[sxoff4[j] + D_IN] = mf2;
      }
    }

    __syncthreads();                              // sX / sXhat ready
    tc += D_IN;
  }
}

extern "C" void kernel_launch(void* const* d_in, const int* in_sizes, int n_in,
                              void* d_out, int out_size, void* d_ws, size_t ws_size,
                              hipStream_t stream) {
  (void)in_sizes; (void)n_in; (void)out_size; (void)d_ws; (void)ws_size;
  const float* x     = (const float*)d_in[0];
  const float* W_ih  = (const float*)d_in[1];
  const float* W_hh  = (const float*)d_in[2];
  const float* b_ih  = (const float*)d_in[3];
  const float* b_hh  = (const float*)d_in[4];
  const float* W_out = (const float*)d_in[5];
  const float* b_out = (const float*)d_in[6];
  const int*   mask  = (const int*)d_in[7];
  float* out = (float*)d_out;

  dim3 grid(B_TOT / BM);   // 32 independent batch tiles
  dim3 block(256);         // 8 wave32s
  rnn_imputer_kernel<<<grid, block, 0, stream>>>(x, W_ih, W_hh, b_ih, b_hh,
                                                 W_out, b_out, mask, out);
}
